// SpectralSimilarityMixer_35450660061472
// MI455X (gfx1250) — matrix operand order repaired and checked by reading, not compiled, not run on hardware
//
#include <hip/hip_runtime.h>

typedef _Float16 h16 __attribute__((ext_vector_type(16)));
typedef _Float16 h8  __attribute__((ext_vector_type(8)));
typedef _Float16 h4  __attribute__((ext_vector_type(4)));
typedef float    f8  __attribute__((ext_vector_type(8)));

constexpr int Bn = 4, Cn = 2048, Dn = 1024, Hn = 8, DKn = 128;
constexpr int BQ = 128;        // query rows per workgroup (8 waves x 16 rows)
constexpr int BK = 64;         // key rows per iteration
constexpr int NWAVE = 8, NTHREADS = 256;
// d_k^-0.5 * log2(e): scores are produced directly in the log2 domain so the
// softmax can use the raw hardware v_exp_f32 (exp2) with no pre-multiply.
constexpr float SCALE = 0.08838834764831845f * 1.44269504088896340736f;

__device__ __forceinline__ float fast_exp2(float x) {
  return __builtin_amdgcn_exp2f(x);
}

// Combine two 16B LDS reads into one 16xf16 WMMA operand.
__device__ __forceinline__ h16 ld_frag(const _Float16* p0, const _Float16* p1) {
  h8 a = *(const h8*)p0;
  h8 b = *(const h8*)p1;
  return __builtin_shufflevector(a, b, 0,1,2,3,4,5,6,7,8,9,10,11,12,13,14,15);
}

// 8 consecutive f32 from global -> 8 x f16 (scaled)
__device__ __forceinline__ h8 cvt8(const float* p, float sc) {
  float4 a = *(const float4*)p;
  float4 b = *(const float4*)(p + 4);
  h8 r = { (_Float16)(a.x*sc), (_Float16)(a.y*sc), (_Float16)(a.z*sc), (_Float16)(a.w*sc),
           (_Float16)(b.x*sc), (_Float16)(b.y*sc), (_Float16)(b.z*sc), (_Float16)(b.w*sc) };
  return r;
}

__global__ __launch_bounds__(NTHREADS)
void SpectralSimilarityMixer_fa_kernel(const float* __restrict__ x,
                                       const float* __restrict__ mix_scale,
                                       float* __restrict__ out) {
  __shared__ __align__(16) _Float16 Ks[BK][DKn];        // 16KB: K rows (row-major)
  __shared__ __align__(16) _Float16 Vt[DKn][BK];        // 16KB: V transposed
  __shared__ __align__(16) _Float16 Ps[NWAVE][16][BK];  // 16KB: per-wave P staging

  const int tid  = threadIdx.x;
  const int lane = tid & 31;
  const int wave = tid >> 5;
  const int hf   = lane >> 4;   // half-wave (0/1)
  const int l16  = lane & 15;

  const int qblk = blockIdx.x;          // 0..15
  const int bh   = blockIdx.y;          // 0..31
  const int b    = bh >> 3;
  const int h    = bh & 7;

  const float ms = mix_scale[0];

  // ---- Q fragments straight from global (A-layout: lane=M, K pairs by half) ----
  const float* xq = x + ((size_t)(b*Cn + qblk*BQ + wave*16 + l16))*Dn + h*DKn;
  h16 aq[4];
#pragma unroll
  for (int c = 0; c < 4; ++c)
    aq[c] = __builtin_shufflevector(cvt8(xq + 32*c + 8*hf,       SCALE),
                                    cvt8(xq + 32*c + 16 + 8*hf,  SCALE),
                                    0,1,2,3,4,5,6,7,8,9,10,11,12,13,14,15);

  f8 o[8];
#pragma unroll
  for (int j = 0; j < 8; ++j) o[j] = f8{0,0,0,0,0,0,0,0};
  float mrow[8], lrow[8];
#pragma unroll
  for (int v = 0; v < 8; ++v) { mrow[v] = -INFINITY; lrow[v] = 0.0f; }

  const float* xk = x + ((size_t)(b*Cn))*Dn + h*DKn;

  for (int kb = 0; kb < Cn/BK; ++kb) {
    __syncthreads();  // previous iteration's readers done with Ks/Vt
    // ---- cooperative load of K/V block: Ks row-major + Vt transposed ----
#pragma unroll
    for (int i = 0; i < (BK*DKn/4)/NTHREADS; ++i) {   // 8 float4 per thread
      int idx = tid + NTHREADS*i;
      int r   = idx >> 5;          // 0..63
      int c4  = (idx & 31) << 2;   // 0..124
      float4 v = *(const float4*)(xk + (size_t)(kb*BK + r)*Dn + c4);
      _Float16 h0 = (_Float16)v.x, h1 = (_Float16)v.y,
               h2 = (_Float16)v.z, h3 = (_Float16)v.w;
      h4 hv = { h0, h1, h2, h3 };
      *(h4*)&Ks[r][c4] = hv;
      Vt[c4+0][r] = h0; Vt[c4+1][r] = h1; Vt[c4+2][r] = h2; Vt[c4+3][r] = h3;
    }
    __syncthreads();

    // ---- S(log2 domain) = (Q*scale*log2e) K^T : 4 tiles, 4 k-chunks each ----
    f8 s[4];
#pragma unroll
    for (int j = 0; j < 4; ++j) {
      f8 acc = f8{0,0,0,0,0,0,0,0};
      const _Float16* krow = &Ks[16*j + l16][0];
#pragma unroll
      for (int c = 0; c < 4; ++c) {
        h16 bk = ld_frag(krow + 32*c + 8*hf, krow + 32*c + 16 + 8*hf);
        acc = __builtin_amdgcn_wmma_f32_16x16x32_f16(false, aq[c], false, bk,
                                                     (short)0, acc, false, false);
      }
      s[j] = acc;
    }

    // ---- online softmax (base-2) over this 16x64 strip ----
    float mnew[8], corr[8];
#pragma unroll
    for (int v = 0; v < 8; ++v) {
      float t = fmaxf(fmaxf(s[0][v], s[1][v]), fmaxf(s[2][v], s[3][v]));
      t = fmaxf(t, __shfl_xor(t, 1));
      t = fmaxf(t, __shfl_xor(t, 2));
      t = fmaxf(t, __shfl_xor(t, 4));
      t = fmaxf(t, __shfl_xor(t, 8));
      mnew[v] = fmaxf(mrow[v], t);
      corr[v] = fast_exp2(mrow[v] - mnew[v]);
      mrow[v] = mnew[v];
    }
#pragma unroll
    for (int j = 0; j < 4; ++j)
#pragma unroll
      for (int v = 0; v < 8; ++v)
        s[j][v] = fast_exp2(s[j][v] - mnew[v]);
#pragma unroll
    for (int v = 0; v < 8; ++v) {
      float t = (s[0][v] + s[1][v]) + (s[2][v] + s[3][v]);
      t += __shfl_xor(t, 1);
      t += __shfl_xor(t, 2);
      t += __shfl_xor(t, 4);
      t += __shfl_xor(t, 8);
      lrow[v] = lrow[v]*corr[v] + t;
    }
#pragma unroll
    for (int j = 0; j < 8; ++j)
#pragma unroll
      for (int v = 0; v < 8; ++v)
        o[j][v] *= corr[v];

    // ---- stage P to per-wave LDS (C-layout -> A-layout conversion) ----
#pragma unroll
    for (int j = 0; j < 4; ++j)
#pragma unroll
      for (int v = 0; v < 8; ++v)
        Ps[wave][v + 8*hf][16*j + l16] = (_Float16)s[j][v];

    h16 ap[2];
    {
      const _Float16* prow = &Ps[wave][l16][0];
#pragma unroll
      for (int c = 0; c < 2; ++c)
        ap[c] = ld_frag(prow + 32*c + 8*hf, prow + 32*c + 16 + 8*hf);
    }

    // ---- O += P V (B operand = V columns, contiguous in Vt) ----
#pragma unroll
    for (int j = 0; j < 8; ++j) {
      const _Float16* vrow = &Vt[16*j + l16][0];
      f8 acc = o[j];
#pragma unroll
      for (int c = 0; c < 2; ++c) {
        h16 bv = ld_frag(vrow + 32*c + 8*hf, vrow + 32*c + 16 + 8*hf);
        acc = __builtin_amdgcn_wmma_f32_16x16x32_f16(false, ap[c], false, bv,
                                                     (short)0, acc, false, false);
      }
      o[j] = acc;
    }
  }

  // ---- epilogue: normalize by row sum, apply mix_scale ----
  const int row0 = qblk*BQ + wave*16;
#pragma unroll
  for (int v = 0; v < 8; ++v) {
    float inv = ms / lrow[v];
    int gr = b*Cn + row0 + v + 8*hf;
    float* orow = out + (size_t)gr*Dn + h*DKn;
#pragma unroll
    for (int j = 0; j < 8; ++j)
      orow[16*j + l16] = o[j][v] * inv;
  }
}

extern "C" void kernel_launch(void* const* d_in, const int* in_sizes, int n_in,
                              void* d_out, int out_size, void* d_ws, size_t ws_size,
                              hipStream_t stream) {
  (void)in_sizes; (void)n_in; (void)out_size; (void)d_ws; (void)ws_size;
  const float* x  = (const float*)d_in[0];
  const float* ms = (const float*)d_in[1];
  float* out      = (float*)d_out;
  dim3 grid(Cn / BQ, Bn * Hn);   // 16 x 32 workgroups
  SpectralSimilarityMixer_fa_kernel<<<grid, NTHREADS, 0, stream>>>(x, ms, out);
}
